// GRUModelNoMLP_87153476370530
// MI455X (gfx1250) — matrix operand order repaired and checked by reading, not compiled, not run on hardware
//
#include <hip/hip_runtime.h>
#include <hip/hip_fp16.h>

#define B_SZ   64
#define T_SZ   2048
#define E_SZ   512
#define H_SZ   32
#define G_SZ   96      // 3*H (gates per direction)
#define NCOL   192     // gates, both directions stacked
#define OUT_SZ 131
#define M_SZ   (B_SZ * T_SZ)

typedef __attribute__((ext_vector_type(16))) _Float16 v16h;
typedef __attribute__((ext_vector_type(8)))  _Float16 v8h;
typedef __attribute__((ext_vector_type(8)))  float    v8f;
typedef __attribute__((ext_vector_type(4)))  float    v4f;

// ---------------------------------------------------------------------------
// Kernel 0: pack W_ih_f / W_ih_b into one f16 matrix [192][512], pack biases.
// ---------------------------------------------------------------------------
__global__ __launch_bounds__(256) void pack_weights(
    const float* __restrict__ wf, const float* __restrict__ wb,
    const float* __restrict__ bf, const float* __restrict__ bb,
    _Float16* __restrict__ wh, float* __restrict__ bias)
{
    int i = blockIdx.x * 256 + threadIdx.x;      // 0 .. 192*512-1
    int r = i >> 9;
    int k = i & 511;
    float v = (r < G_SZ) ? wf[r * E_SZ + k] : wb[(r - G_SZ) * E_SZ + k];
    wh[i] = (_Float16)v;
    if (i < NCOL) bias[i] = (i < G_SZ) ? bf[i] : bb[i - G_SZ];
}

// ---------------------------------------------------------------------------
// Kernel 1: xp[b*T+t][g] = x[b][t][:] . W_ih[g][:] + b_ih[g]   (g = 0..191)
// Block = 256 threads = 8 waves. Each wave: 16 rows x 192 cols, K=512.
// WMMA f32_16x16x32_f16, 12 N-tiles held in 96 accumulator VGPRs.
// ---------------------------------------------------------------------------
__global__ __launch_bounds__(256) void gemm_xproj(
    const float* __restrict__ x, const _Float16* __restrict__ wh,
    const float* __restrict__ bias, float* __restrict__ xp)
{
    const int wave  = threadIdx.x >> 5;
    const int lane  = threadIdx.x & 31;
    const int mbase = blockIdx.x * 128 + wave * 16;
    const int hrow  = lane & 15;            // A: M row / B: N col / D: N col
    const int kb    = (lane >> 4) * 8;      // A-fragment K sub-offset
    const int kst   = (lane >> 4) * 16;     // unused; B fragment also uses kb

    const float* xrow = x + (size_t)(mbase + hrow) * E_SZ;

    v8f acc[12] = {};

    for (int s = 0; s < 16; ++s) {
        const int k0 = s * 32;
        if (s + 2 < 16)
            __builtin_prefetch(xrow + k0 + 64, 0, 1);

        // ---- A fragment: 16x32 f16, converted from f32 in registers ----
        v4f a0 = *reinterpret_cast<const v4f*>(xrow + k0 + kb);
        v4f a1 = *reinterpret_cast<const v4f*>(xrow + k0 + kb + 4);
        v4f a2 = *reinterpret_cast<const v4f*>(xrow + k0 + 16 + kb);
        v4f a3 = *reinterpret_cast<const v4f*>(xrow + k0 + 16 + kb + 4);
        v16h A;
#pragma unroll
        for (int i = 0; i < 4; ++i) {
            A[i]      = (_Float16)a0[i];
            A[4 + i]  = (_Float16)a1[i];
            A[8 + i]  = (_Float16)a2[i];
            A[12 + i] = (_Float16)a3[i];
        }

        // ---- 12 N-tiles: load f16 B fragment, WMMA accumulate ----
#pragma unroll
        for (int tt = 0; tt < 12; ++tt) {
            const _Float16* wr = wh + (size_t)(tt * 16 + hrow) * E_SZ + k0 + kb;
            v8h b0 = *reinterpret_cast<const v8h*>(wr);
            v8h b1 = *reinterpret_cast<const v8h*>(wr + 16);
            v16h Bv = __builtin_shufflevector(b0, b1,
                        0, 1, 2, 3, 4, 5, 6, 7, 8, 9, 10, 11, 12, 13, 14, 15);
            acc[tt] = __builtin_amdgcn_wmma_f32_16x16x32_f16(
                false, A, false, Bv, (short)0, acc[tt], false, false);
        }
    }

    // ---- Epilogue: D layout (lane n=hrow, rows m = (lane>>4)*8 + r) ----
    const int moff = (lane >> 4) * 8;
#pragma unroll
    for (int tt = 0; tt < 12; ++tt) {
        const int g  = tt * 16 + hrow;
        const float bv = bias[g];
#pragma unroll
        for (int r = 0; r < 8; ++r) {
            const size_t row = (size_t)(mbase + moff + r);
            xp[row * NCOL + g] = acc[tt][r] + bv;
        }
    }
}

// ---------------------------------------------------------------------------
// Kernel 2: GRU scan. One wave per (batch, direction); lane j holds h[j].
// Matvec h@W_hh.T via v_readlane broadcast; gates are lane-parallel.
// Masked steps are identity -> iterate exactly len[b] steps, no masking.
// ---------------------------------------------------------------------------
__global__ __launch_bounds__(32) void gru_scan(
    const float* __restrict__ xp, const int* __restrict__ src_len,
    const float* __restrict__ Whf, const float* __restrict__ bhf,
    const float* __restrict__ Whb, const float* __restrict__ bhb,
    float* __restrict__ hid)
{
    const int b    = blockIdx.x & (B_SZ - 1);
    const int dir  = blockIdx.x >> 6;
    const int lane = threadIdx.x;

    const float* Wh = dir ? Whb : Whf;
    const float* bh = dir ? bhb : bhf;

    float wr[H_SZ], wz[H_SZ], wn[H_SZ];
#pragma unroll
    for (int k = 0; k < H_SZ; ++k) {
        wr[k] = Wh[(lane)          * H_SZ + k];
        wz[k] = Wh[(lane + H_SZ)   * H_SZ + k];
        wn[k] = Wh[(lane + 2*H_SZ) * H_SZ + k];
    }
    const float br = bh[lane];
    const float bz = bh[lane + H_SZ];
    const float bn = bh[lane + 2 * H_SZ];

    int len = src_len[b];
    len = len < 0 ? 0 : (len > T_SZ ? T_SZ : len);

    const float* xbase = xp + (size_t)b * T_SZ * NCOL + dir * G_SZ + lane;
    const int t0   = dir ? (len - 1) : 0;
    const int step = dir ? -1 : 1;

    float h = 0.0f;
    for (int i = 0; i < len; ++i) {
        const int t = t0 + step * i;
        const float* xg = xbase + (size_t)t * NCOL;
        const float xr = xg[0];
        const float xz = xg[H_SZ];
        const float xn = xg[2 * H_SZ];

        float a0 = br, a1 = bz, a2 = bn;
#pragma unroll
        for (int k = 0; k < H_SZ; ++k) {
            const float hk = __uint_as_float(
                __builtin_amdgcn_readlane(__float_as_uint(h), k));
            a0 = fmaf(wr[k], hk, a0);
            a1 = fmaf(wz[k], hk, a1);
            a2 = fmaf(wn[k], hk, a2);
        }

        const float r  = __builtin_amdgcn_rcpf(1.0f + __expf(-(xr + a0)));
        const float z  = __builtin_amdgcn_rcpf(1.0f + __expf(-(xz + a1)));
        const float nx = xn + r * a2;
        const float e2 = __expf(-2.0f * nx);
        const float nn = (1.0f - e2) * __builtin_amdgcn_rcpf(1.0f + e2);
        h = fmaf(z, h - nn, nn);           // (1-z)*n + z*h
    }

    hid[b * (2 * H_SZ) + dir * H_SZ + lane] = h;
}

// ---------------------------------------------------------------------------
// Kernel 3: out[b][o] = hidden[b][:] . fc_W[o][:] + fc_b[o]
// ---------------------------------------------------------------------------
__global__ __launch_bounds__(160) void fc_kernel(
    const float* __restrict__ hid, const float* __restrict__ fcW,
    const float* __restrict__ fcb, float* __restrict__ out)
{
    const int b = blockIdx.x;
    const int o = threadIdx.x;
    __shared__ float hs[2 * H_SZ];
    if (threadIdx.x < 2 * H_SZ) hs[threadIdx.x] = hid[b * (2 * H_SZ) + threadIdx.x];
    __syncthreads();
    if (o < OUT_SZ) {
        float s = fcb[o];
#pragma unroll
        for (int c = 0; c < 2 * H_SZ; ++c)
            s = fmaf(hs[c], fcW[o * (2 * H_SZ) + c], s);
        out[b * OUT_SZ + o] = s;
    }
}

// ---------------------------------------------------------------------------
extern "C" void kernel_launch(void* const* d_in, const int* in_sizes, int n_in,
                              void* d_out, int out_size, void* d_ws, size_t ws_size,
                              hipStream_t stream)
{
    const float* x      = (const float*)d_in[0];
    const int*   slen   = (const int*)  d_in[1];
    // d_in[2] = other_input (unused by the reference output)
    const float* W_ih_f = (const float*)d_in[3];
    const float* W_hh_f = (const float*)d_in[4];
    const float* b_ih_f = (const float*)d_in[5];
    const float* b_hh_f = (const float*)d_in[6];
    const float* W_ih_b = (const float*)d_in[7];
    const float* W_hh_b = (const float*)d_in[8];
    const float* b_ih_b = (const float*)d_in[9];
    const float* b_hh_b = (const float*)d_in[10];
    const float* fcW    = (const float*)d_in[11];
    const float* fcb    = (const float*)d_in[12];
    float* out = (float*)d_out;

    char* ws = (char*)d_ws;
    _Float16* wh   = (_Float16*)ws;                    // 192*512*2   = 196608 B
    float*    bias = (float*)(ws + 196608);            // 192*4       = 768 B
    float*    hid  = (float*)(ws + 197376);            // 64*64*4     = 16384 B
    float*    xp   = (float*)(ws + 262144);            // 131072*192*4 = 96 MiB

    pack_weights<<<(NCOL * E_SZ) / 256, 256, 0, stream>>>(
        W_ih_f, W_ih_b, b_ih_f, b_ih_b, wh, bias);

    gemm_xproj<<<M_SZ / 128, 256, 0, stream>>>(x, wh, bias, xp);

    gru_scan<<<2 * B_SZ, 32, 0, stream>>>(
        xp, slen, W_hh_f, b_hh_f, W_hh_b, b_hh_b, hid);

    fc_kernel<<<B_SZ, 160, 0, stream>>>(hid, fcW, fcb, out);
}
